// RelativeGPT_6425271075112
// MI455X (gfx1250) — matrix operand order repaired and checked by reading, compiled, not run on hardware
//
#include <hip/hip_runtime.h>
#include <hip/hip_bf16.h>
#include <math.h>

// Model dims (fixed by reference)
#define Bb 2
#define Ss 1024
#define Dd 1024
#define Hh 16
#define Ll 6
#define Vv 32000
#define Ff 4096
#define Mm 1024
#define Dh 64

typedef __attribute__((ext_vector_type(16))) __bf16          v16bf;
typedef __attribute__((ext_vector_type(16))) unsigned short  v16u;
typedef __attribute__((ext_vector_type(8)))  float           v8f;

struct U4x2 { uint4 a, b; };

__device__ __forceinline__ unsigned short f2bf(float f) {
  unsigned int u = __builtin_bit_cast(unsigned int, f);
  unsigned int r = (u + 0x7FFFu + ((u >> 16) & 1u)) >> 16;  // RNE
  return (unsigned short)r;
}
__device__ __forceinline__ v16bf asbf(v16u v) { return __builtin_bit_cast(v16bf, v); }
__device__ __forceinline__ v16u  mkfrag(uint4 lo, uint4 hi) {
  U4x2 p{lo, hi};
  return __builtin_bit_cast(v16u, p);
}
// A fragment: contiguous 32B (K 0..15 / 16..31 per half-wave)
__device__ __forceinline__ v16u ldA(const unsigned short* p) {
  return mkfrag(*(const uint4*)p, *(const uint4*)(p + 8));
}
// B fragment: contiguous 32B (K 0..15 / 16..31 per half-wave)
__device__ __forceinline__ v16u ldB(const unsigned short* p) {
  return mkfrag(*(const uint4*)p, *(const uint4*)(p + 8));
}

__device__ __forceinline__ v8f wmma_bf16(v16u a, v16u b, v8f c) {
  // D = A(16x32 bf16) * B(32x16 bf16) + C(16x16 f32)
  return __builtin_amdgcn_wmma_f32_16x16x32_bf16(false, asbf(a), false, asbf(b),
                                                 (short)0, c, false, false);
}

// wave-local LDS producer->consumer ordering (no cross-wave dependency)
__device__ __forceinline__ void wave_lds_sync() {
  __builtin_amdgcn_wave_barrier();
  asm volatile("s_wait_dscnt 0" ::: "memory");
  __builtin_amdgcn_wave_barrier();
}

__device__ __forceinline__ float grp16_max(float v) {
  #pragma unroll
  for (int m = 1; m < 16; m <<= 1) v = fmaxf(v, __shfl_xor(v, m, 16));
  return v;
}
__device__ __forceinline__ float grp16_sum(float v) {
  #pragma unroll
  for (int m = 1; m < 16; m <<= 1) v += __shfl_xor(v, m, 16);
  return v;
}

// ---------------------------------------------------------------------------
// fp32 [K][N] weight  ->  bf16 [N][K] (transposed) via LDS tile
// ---------------------------------------------------------------------------
__global__ __launch_bounds__(256) void cvt_transpose_kernel(
    const float* __restrict__ in, unsigned short* __restrict__ out,
    int Kdim, int Ndim) {
  __shared__ float t[32][33];
  const int nb = blockIdx.x * 32, kb = blockIdx.y * 32;
  const int tx = threadIdx.x & 31, ty = threadIdx.x >> 5;  // 32 x 8
  #pragma unroll
  for (int j = 0; j < 4; ++j)
    t[ty + j * 8][tx] = in[(size_t)(kb + ty + j * 8) * Ndim + nb + tx];
  __syncthreads();
  #pragma unroll
  for (int j = 0; j < 4; ++j)
    out[(size_t)(nb + ty + j * 8) * Kdim + kb + tx] = f2bf(t[tx][ty + j * 8]);
}

// plain fp32 -> bf16 (row-major preserved), 4 elems / thread
__global__ __launch_bounds__(256) void cvt_kernel(const float* __restrict__ in,
                                                  unsigned short* __restrict__ out,
                                                  int n) {
  const int i = (blockIdx.x * 256 + threadIdx.x) * 4;
  if (i + 3 < n) {
    #pragma unroll
    for (int j = 0; j < 4; ++j) out[i + j] = f2bf(in[i + j]);
  }
}

// ---------------------------------------------------------------------------
// Embedding gather: writes f32 residual stream + bf16 shadow
// ---------------------------------------------------------------------------
__global__ __launch_bounds__(256) void embed_kernel(const int* __restrict__ tok,
                                                    const float* __restrict__ emb,
                                                    float* __restrict__ x,
                                                    unsigned short* __restrict__ x16) {
  const int row = blockIdx.x;
  const int t = tok[row];
  #pragma unroll
  for (int j = 0; j < 4; ++j) {
    const int c = threadIdx.x * 4 + j;
    const float v = emb[(size_t)t * Dd + c];
    x[(size_t)row * Dd + c] = v;
    x16[(size_t)row * Dd + c] = f2bf(v);
  }
}

// ---------------------------------------------------------------------------
// bf16 WMMA GEMM:  C[M,N] = act(A16[M,K] @ Wt16[N,K]^T + bias[N])
// 256 thr = 8 waves; each wave: 16(M) x 64(N) tile = 4 accumulators.
// K-loop unrolled by 64 with two named fragment sets: each half issues the
// other half's loads before its 4 wmmas (latency hidden, no register copies,
// fixed-base + offset addressing keeps loads on the global path).
// Requires Kdim % 64 == 0 (true for 1024 and 4096).
// out_mode: 0 = f32 row-major, 1 = bf16 row-major, 2 = bf16 transposed [N][M]
// ---------------------------------------------------------------------------
__global__ __launch_bounds__(256) void gemm_bf16_kernel(
    const unsigned short* __restrict__ A, const unsigned short* __restrict__ Wt,
    const float* __restrict__ bias, float* __restrict__ Cf,
    unsigned short* __restrict__ Ch, int Mdim, int Ndim, int Kdim,
    int act, int out_mode) {
  const int tid  = threadIdx.x;
  const int lane = tid & 31, wid = tid >> 5;
  const int half = lane >> 4, lrow = lane & 15;
  const int m0 = blockIdx.y * 128 + wid * 16;    // wave's 16 rows
  const int n_blk = blockIdx.x * 64;             // wave's 64 cols

  const v8f vz = {0.f, 0.f, 0.f, 0.f, 0.f, 0.f, 0.f, 0.f};
  v8f acc[4] = {vz, vz, vz, vz};

  const unsigned short* aB  = A + (size_t)(m0 + lrow) * Kdim + half * 16;
  const unsigned short* bB0 = Wt + (size_t)(n_blk + lrow) * Kdim + half * 16;
  const unsigned short* bB1 = bB0 + (size_t)16 * Kdim;
  const unsigned short* bB2 = bB0 + (size_t)32 * Kdim;
  const unsigned short* bB3 = bB0 + (size_t)48 * Kdim;

  v16u a0  = ldA(aB);
  v16u b00 = ldB(bB0), b01 = ldB(bB1), b02 = ldB(bB2), b03 = ldB(bB3);

  for (int kb = 0; kb < Kdim; kb += 64) {
    // slab kb+32 into set 1 (always in range: Kdim % 64 == 0)
    const v16u a1  = ldA(aB + kb + 32);
    const v16u b10 = ldB(bB0 + kb + 32), b11 = ldB(bB1 + kb + 32),
               b12 = ldB(bB2 + kb + 32), b13 = ldB(bB3 + kb + 32);
    __builtin_prefetch(aB + kb + 128, 0, 1);

    acc[0] = wmma_bf16(a0, b00, acc[0]);
    acc[1] = wmma_bf16(a0, b01, acc[1]);
    acc[2] = wmma_bf16(a0, b02, acc[2]);
    acc[3] = wmma_bf16(a0, b03, acc[3]);

    if (kb + 64 < Kdim) {                        // slab kb+64 back into set 0
      a0  = ldA(aB + kb + 64);
      b00 = ldB(bB0 + kb + 64); b01 = ldB(bB1 + kb + 64);
      b02 = ldB(bB2 + kb + 64); b03 = ldB(bB3 + kb + 64);
    }
    acc[0] = wmma_bf16(a1, b10, acc[0]);
    acc[1] = wmma_bf16(a1, b11, acc[1]);
    acc[2] = wmma_bf16(a1, b12, acc[2]);
    acc[3] = wmma_bf16(a1, b13, acc[3]);
  }

  // C layout: VGPR i -> row m0+i (lanes<16) / m0+8+i (lanes>=16), col lane%16
  #pragma unroll
  for (int c = 0; c < 4; ++c)
    #pragma unroll
    for (int i = 0; i < 8; ++i) {
      const int row = m0 + half * 8 + i;
      const int col = n_blk + c * 16 + lrow;
      float v = acc[c][i] + bias[col];
      if (act == 1) {  // tanh-approx GELU (jax.nn.gelu default)
        const float x3 = v * v * v;
        v = 0.5f * v * (1.f + tanhf(0.7978845608028654f * (v + 0.044715f * x3)));
      }
      if (out_mode == 0)      Cf[(size_t)row * Ndim + col] = v;
      else if (out_mode == 1) Ch[(size_t)row * Ndim + col] = f2bf(v);
      else                    Ch[(size_t)col * Mdim + row] = f2bf(v);  // [N][M]
    }
}

// ---------------------------------------------------------------------------
// Flash attention with Music-Transformer relative skew (all-bf16 inputs).
//   logits[q,k] = (q.k + q.rpe[S-1-(q-k)]) / sqrt(Dh), causal
// Q,K: bf16 [B*S][D] (head-sliced); Vt: bf16 [D][B*S] (transposed by GEMM);
// E: bf16 [M][Dh] per layer; O: bf16 [B*S][D].
// Block = 128 thr = 4 waves; each wave owns 16 q-rows; 32-key tiles.
// ---------------------------------------------------------------------------
__global__ __launch_bounds__(128) void attn_kernel(
    const unsigned short* __restrict__ Q, const unsigned short* __restrict__ K,
    const unsigned short* __restrict__ Vt, const unsigned short* __restrict__ E,
    unsigned short* __restrict__ O) {
  __shared__ float          Tlds[4][16 * 48];   // per-wave srel scratch
  __shared__ unsigned short Plds[4][16 * 32];   // per-wave P repack (bf16)

  const int tid  = threadIdx.x;
  const int lane = tid & 31, wid = tid >> 5;
  const int half = lane >> 4, lrow = lane & 15;
  const int h = blockIdx.y, b = blockIdx.z;
  const int q0 = blockIdx.x * 64 + wid * 16;

  // Preload Q A-fragments for both 32-wide d-chunks (Dh = 64)
  const unsigned short* qrow = Q + (size_t)(b * Ss + q0 + lrow) * Dd + h * Dh;
  v16u qf[2];
  #pragma unroll
  for (int c = 0; c < 2; ++c) qf[c] = ldA(qrow + c * 32 + half * 16);

  const v8f vz = {0.f, 0.f, 0.f, 0.f, 0.f, 0.f, 0.f, 0.f};
  v8f oacc[4] = {vz, vz, vz, vz};               // 16 x 64 output accumulator
  float mrow[8], lsum[8];
  #pragma unroll
  for (int i = 0; i < 8; ++i) { mrow[i] = -1e30f; lsum[i] = 0.f; }

  const int nkt = (q0 + 15) / 32 + 1;           // causal key-tile count
  for (int kt = 0; kt < nkt; ++kt) {
    const int k0 = kt * 32;

    // ---- S = Q K^T over 16x32 tile (2 N-chunks x 2 K-chunks) ----
    v8f sc[2] = {vz, vz};
    #pragma unroll
    for (int c = 0; c < 2; ++c)
      #pragma unroll
      for (int kc = 0; kc < 2; ++kc) {
        const unsigned short* kr =
            K + (size_t)(b * Ss + k0 + c * 16 + lrow) * Dd + h * Dh + kc * 32 + half * 16;
        sc[c] = wmma_bf16(qf[kc], ldB(kr), sc[c]);
      }

    // ---- T = Q @ E_slab^T : 48 rpe rows cover m = S-1-q+k for this tile ----
    const int m_base = Ss - 16 - q0 + k0;
    v8f tc[3] = {vz, vz, vz};
    #pragma unroll
    for (int mc = 0; mc < 3; ++mc)
      #pragma unroll
      for (int kc = 0; kc < 2; ++kc) {
        int m = m_base + mc * 16 + lrow;
        m = m < 0 ? 0 : (m > Mm - 1 ? Mm - 1 : m);  // clamped rows masked later
        const unsigned short* er = E + (size_t)m * Dh + kc * 32 + half * 16;
        tc[mc] = wmma_bf16(qf[kc], ldB(er), tc[mc]);
      }
    #pragma unroll
    for (int mc = 0; mc < 3; ++mc)
      #pragma unroll
      for (int i = 0; i < 8; ++i)
        Tlds[wid][(i + half * 8) * 48 + mc * 16 + lrow] = tc[mc][i];
    wave_lds_sync();

    // ---- logits + online softmax (per q-row = per accumulator index) ----
    #pragma unroll
    for (int i = 0; i < 8; ++i) {
      const int qi = q0 + i + half * 8;
      float lg[2];
      #pragma unroll
      for (int c = 0; c < 2; ++c) {
        const int key = k0 + c * 16 + lrow;
        const int idx = (q0 + 15 - qi) + (key - k0);            // skew gather
        const float srel = Tlds[wid][(qi - q0) * 48 + idx];
        const float v = (sc[c][i] + srel) * 0.125f;             // 1/sqrt(64)
        lg[c] = (key > qi) ? -1e9f : v;
      }
      const float tmax = grp16_max(fmaxf(lg[0], lg[1]));
      const float mn   = fmaxf(mrow[i], tmax);
      const float scl  = __expf(mrow[i] - mn);
      const float p0 = __expf(lg[0] - mn), p1 = __expf(lg[1] - mn);
      lsum[i] = lsum[i] * scl + grp16_sum(p0 + p1);
      mrow[i] = mn;
      #pragma unroll
      for (int n = 0; n < 4; ++n) oacc[n][i] *= scl;
      Plds[wid][(i + half * 8) * 32 + lrow]      = f2bf(p0);
      Plds[wid][(i + half * 8) * 32 + 16 + lrow] = f2bf(p1);
    }
    wave_lds_sync();

    // ---- P repacked to A-fragment layout (vector LDS reads), O += P @ V ----
    const v16u pf = mkfrag(*(const uint4*)&Plds[wid][lrow * 32 + half * 16],
                           *(const uint4*)&Plds[wid][lrow * 32 + half * 16 + 8]);
    #pragma unroll
    for (int n = 0; n < 4; ++n) {
      const unsigned short* vr =
          Vt + (size_t)(h * Dh + n * 16 + lrow) * (Bb * Ss) + (size_t)b * Ss + k0 + half * 16;
      oacc[n] = wmma_bf16(pf, ldB(vr), oacc[n]);
    }
  }

  #pragma unroll
  for (int n = 0; n < 4; ++n)
    #pragma unroll
    for (int i = 0; i < 8; ++i) {
      const int qi = q0 + i + half * 8;
      O[(size_t)(b * Ss + qi) * Dd + h * Dh + n * 16 + lrow] = f2bf(oacc[n][i] / lsum[i]);
    }
}

// ---------------------------------------------------------------------------
// x = LayerNorm(x + r) * s + b   (in place on x) + bf16 shadow write
// ---------------------------------------------------------------------------
__global__ __launch_bounds__(256) void ln_residual_kernel(
    float* __restrict__ x, const float* __restrict__ r,
    const float* __restrict__ s, const float* __restrict__ bi,
    unsigned short* __restrict__ x16) {
  __shared__ float red[256];
  const int row = blockIdx.x, tid = threadIdx.x;
  float v[4];
  float acc = 0.f;
  #pragma unroll
  for (int j = 0; j < 4; ++j) {
    const int c = tid * 4 + j;
    v[j] = x[(size_t)row * Dd + c] + r[(size_t)row * Dd + c];
    acc += v[j];
  }
  red[tid] = acc;
  __syncthreads();
  for (int off = 128; off > 0; off >>= 1) {
    if (tid < off) red[tid] += red[tid + off];
    __syncthreads();
  }
  const float mu = red[0] * (1.f / Dd);
  __syncthreads();
  acc = 0.f;
  #pragma unroll
  for (int j = 0; j < 4; ++j) { const float d = v[j] - mu; acc += d * d; }
  red[tid] = acc;
  __syncthreads();
  for (int off = 128; off > 0; off >>= 1) {
    if (tid < off) red[tid] += red[tid + off];
    __syncthreads();
  }
  const float inv = rsqrtf(red[0] * (1.f / Dd) + 1e-5f);
  #pragma unroll
  for (int j = 0; j < 4; ++j) {
    const int c = tid * 4 + j;
    const float o = (v[j] - mu) * inv * s[c] + bi[c];
    x[(size_t)row * Dd + c] = o;
    x16[(size_t)row * Dd + c] = f2bf(o);
  }
}

// ---------------------------------------------------------------------------
// Host orchestration
// ---------------------------------------------------------------------------
extern "C" void kernel_launch(void* const* d_in, const int* in_sizes, int n_in,
                              void* d_out, int out_size, void* d_ws, size_t ws_size,
                              hipStream_t stream) {
  (void)in_sizes; (void)n_in; (void)out_size; (void)ws_size;
  const int*   tokens = (const int*)  d_in[0];
  const float* emb    = (const float*)d_in[1];
  const float* Wq     = (const float*)d_in[2];
  const float* bq     = (const float*)d_in[3];
  const float* Wk     = (const float*)d_in[4];
  const float* bk     = (const float*)d_in[5];
  const float* Wv     = (const float*)d_in[6];
  const float* bv     = (const float*)d_in[7];
  const float* Wo     = (const float*)d_in[8];
  const float* bo     = (const float*)d_in[9];
  const float* rpe    = (const float*)d_in[10];
  const float* ln1_s  = (const float*)d_in[11];
  const float* ln1_b  = (const float*)d_in[12];
  const float* ln2_s  = (const float*)d_in[13];
  const float* ln2_b  = (const float*)d_in[14];
  const float* W1     = (const float*)d_in[15];
  const float* b1     = (const float*)d_in[16];
  const float* W2     = (const float*)d_in[17];
  const float* b2     = (const float*)d_in[18];
  const float* Wout   = (const float*)d_in[19];
  const float* bout   = (const float*)d_in[20];
  float* out = (float*)d_out;

  const size_t NTOK = (size_t)Bb * Ss;          // 2048 rows
  const size_t NACT = NTOK * Dd;                // 2,097,152 elems

  // workspace layout
  float* xbuf = (float*)d_ws;                   // residual stream (f32)
  float* pb   = xbuf + NACT;                    // projection scratch (f32)
  unsigned short* xb16 = (unsigned short*)(pb + NACT);
  unsigned short* qb   = xb16 + NACT;
  unsigned short* kb   = qb + NACT;
  unsigned short* vt   = kb + NACT;             // V transposed [D][B*S]
  unsigned short* ob   = vt + NACT;             // attn out (bf16)
  unsigned short* fb   = ob + NACT;             // FFN hidden (bf16, 2048x4096)
  unsigned short* eb   = fb + NTOK * Ff;        // rpe bf16 (L*M*Dh)
  unsigned short* wqT  = eb + (size_t)Ll * Mm * Dh;
  unsigned short* wkT  = wqT + (size_t)Dd * Dd;
  unsigned short* wvT  = wkT + (size_t)Dd * Dd;
  unsigned short* woT  = wvT + (size_t)Dd * Dd;
  unsigned short* w1T  = woT + (size_t)Dd * Dd;       // [F][D]
  unsigned short* w2T  = w1T + (size_t)Dd * Ff;       // [D][F]
  unsigned short* wouT = w2T + (size_t)Dd * Ff;       // [V][D]

  const int MR = (int)NTOK;
  const dim3 gD(Dd / 64, MR / 128);             // N=1024 gemms
  const dim3 gF(Ff / 64, MR / 128);             // N=4096 gemm
  const dim3 gV(Vv / 64, MR / 128);             // N=32000 gemm
  const dim3 gattn(Ss / 64, Hh, Bb);
  const dim3 tDD(Dd / 32, Dd / 32);             // transpose grids
  const dim3 tDF(Ff / 32, Dd / 32);
  const dim3 tFD(Dd / 32, Ff / 32);
  const dim3 tDV(Vv / 32, Dd / 32);

  embed_kernel<<<(int)NTOK, 256, 0, stream>>>(tokens, emb, xbuf, xb16);
  cvt_kernel<<<(Ll * Mm * Dh) / 1024, 256, 0, stream>>>(rpe, eb, Ll * Mm * Dh);
  cvt_transpose_kernel<<<tDV, 256, 0, stream>>>(Wout, wouT, Dd, Vv);

  for (int l = 0; l < Ll; ++l) {
    const size_t wDD = (size_t)l * Dd * Dd, vD = (size_t)l * Dd;
    cvt_transpose_kernel<<<tDD, 256, 0, stream>>>(Wq + wDD, wqT, Dd, Dd);
    cvt_transpose_kernel<<<tDD, 256, 0, stream>>>(Wk + wDD, wkT, Dd, Dd);
    cvt_transpose_kernel<<<tDD, 256, 0, stream>>>(Wv + wDD, wvT, Dd, Dd);
    cvt_transpose_kernel<<<tDD, 256, 0, stream>>>(Wo + wDD, woT, Dd, Dd);
    cvt_transpose_kernel<<<tDF, 256, 0, stream>>>(W1 + (size_t)l * Dd * Ff, w1T, Dd, Ff);
    cvt_transpose_kernel<<<tFD, 256, 0, stream>>>(W2 + (size_t)l * Dd * Ff, w2T, Ff, Dd);

    gemm_bf16_kernel<<<gD, 256, 0, stream>>>(xb16, wqT, bq + vD, nullptr, qb,
                                             MR, Dd, Dd, 0, 1);
    gemm_bf16_kernel<<<gD, 256, 0, stream>>>(xb16, wkT, bk + vD, nullptr, kb,
                                             MR, Dd, Dd, 0, 1);
    gemm_bf16_kernel<<<gD, 256, 0, stream>>>(xb16, wvT, bv + vD, nullptr, vt,
                                             MR, Dd, Dd, 0, 2);       // transposed out

    attn_kernel<<<gattn, 128, 0, stream>>>(qb, kb, vt, eb + (size_t)l * Mm * Dh, ob);

    gemm_bf16_kernel<<<gD, 256, 0, stream>>>(ob, woT, bo + vD, pb, nullptr,
                                             MR, Dd, Dd, 0, 0);
    ln_residual_kernel<<<(int)NTOK, 256, 0, stream>>>(xbuf, pb, ln1_s + vD, ln1_b + vD, xb16);

    gemm_bf16_kernel<<<gF, 256, 0, stream>>>(xb16, w1T, b1 + (size_t)l * Ff, nullptr, fb,
                                             MR, Ff, Dd, 1, 1);       // GELU
    gemm_bf16_kernel<<<gD, 256, 0, stream>>>(fb, w2T, b2 + vD, pb, nullptr,
                                             MR, Dd, Ff, 0, 0);
    ln_residual_kernel<<<(int)NTOK, 256, 0, stream>>>(xbuf, pb, ln2_s + vD, ln2_b + vD, xb16);
  }

  gemm_bf16_kernel<<<gV, 256, 0, stream>>>(xb16, wouT, bout, out, nullptr,
                                           MR, Vv, Dd, 0, 0);
}
